// DeformConv1D_35459249996102
// MI455X (gfx1250) — matrix operand order repaired and checked by reading, not compiled, not run on hardware
//
#include <hip/hip_runtime.h>
#include <hip/hip_bf16.h>

// Deformable Conv1D for MI455X (gfx1250, wave32, WMMA).
// Main conv == GEMM M=128, K=192, N=B*h=262144 -> 12.9 GFLOP.
// bf16 WMMA (16x16x32) with f32 accumulate; B-panel gathered into LDS in
// WMMA fragment layout; A (weights) pre-packed to fragment layout in ws.

typedef __attribute__((ext_vector_type(16))) __bf16 v16bf;
typedef __attribute__((ext_vector_type(8)))  float  v8f;

#define B_   64
#define C_   64
#define L_   4096
#define OC_  128
#define KS_  3
#define H_   4096
#define W_   (L_ + 2)          // padded width 4098
#define KTOT (C_ * KS_)        // 192 = GEMM K

// workspace layout: [params: B*H*KS float4][packed bf16 weights]
#define PARAM_FLOATS ((size_t)B_ * H_ * KS_ * 4)

static __device__ __forceinline__ unsigned short f32_bf16(float f) {
    unsigned u = __float_as_uint(f);
    u += 0x7FFFu + ((u >> 16) & 1u);       // round-to-nearest-even
    return (unsigned short)(u >> 16);
}

// ---- Kernel 0: pack w_conv (128,64,3) f32 -> bf16 A-fragment layout ----
// 16-bit A 16x32 layout (ISA 7.12.2): lane = M&15 (+16 if K's upper-8 of its
// 16-group), VGPR v: first 4 hold K 0..15, last 4 hold K 16..31, 2 K per VGPR.
__global__ void pack_w(const float* __restrict__ wconv,
                       unsigned short* __restrict__ wp) {
    int t = blockIdx.x * blockDim.x + threadIdx.x;
    if (t >= OC_ * KTOT) return;
    int oc = t / KTOT, K = t % KTOT;
    int m = oc >> 4, laneLo = oc & 15;
    int kk = K >> 5, k32 = K & 31;
    int hi16  = (k32 >> 4) & 1;            // K in upper 16 of 32 -> VGPR 4..7
    int w16   = k32 & 15;
    int half8 = (w16 >> 3) & 1;            // upper 8 of 16 -> lanes 16..31
    int rem   = w16 & 7;
    int v = hi16 * 4 + (rem >> 1);
    int j = rem & 1;
    int lane = laneLo + half8 * 16;
    int e = 2 * v + j;                     // element within v16bf
    wp[(((m * 6 + kk) * 32) + lane) * 16 + e] = f32_bf16(wconv[t]);
}

// ---- Kernel 1: offset conv + sampling params ----
// params[(b*H+i)*KS+k] = float4{ g_lt, g_rb, bitcast(i_lt), bitcast(i_rb) }
// indices into UNPADDED x; pad taps get weight 0 (xp pad is zero).
__global__ __launch_bounds__(128) void offs_k(const float* __restrict__ x,
                                              const float* __restrict__ woff,
                                              const float* __restrict__ boff,
                                              float* __restrict__ params) {
    __shared__ float xs[C_][130];          // x[b, :, i0-1 .. i0+128]
    __shared__ float wl[KS_ * C_ * 3];
    __shared__ float bl[KS_];
    int b = blockIdx.x >> 5;
    int tile = blockIdx.x & 31;
    int i0 = tile * 128;
    int tid = threadIdx.x;

    for (int q = tid; q < KS_ * C_ * 3; q += 128) wl[q] = woff[q];
    if (tid < KS_) bl[tid] = boff[tid];
    for (int q = tid; q < C_ * 130; q += 128) {
        int c = q / 130, pp = q % 130;
        int gi = i0 - 1 + pp;
        xs[c][pp] = (gi >= 0 && gi < L_) ? x[((size_t)b * C_ + c) * L_ + gi] : 0.f;
    }
    __syncthreads();

    int i = i0 + tid;
    float off[KS_];
    #pragma unroll
    for (int k = 0; k < KS_; ++k) {
        float acc = bl[k];
        for (int c = 0; c < C_; ++c) {
            const float* wr = &wl[(k * C_ + c) * 3];
            acc += wr[0] * xs[c][tid] + wr[1] * xs[c][tid + 1] + wr[2] * xs[c][tid + 2];
        }
        off[k] = acc;
    }
    #pragma unroll
    for (int k = 0; k < KS_; ++k) {
        float p  = (float)(i + k) + off[k];          // p0 + p_n + offset
        float fl = floorf(p);
        float qlt = fminf(fmaxf(fl,       0.f), (float)(W_ - 1));
        float qrb = fminf(fmaxf(fl + 1.f, 0.f), (float)(W_ - 1));
        float mask = (p < 1.f || p > (float)(W_ - 2)) ? 1.f : 0.f;
        p = p * (1.f - mask) + fl * mask;
        p = fminf(fmaxf(p, 0.f), (float)(W_ - 1));
        float g = (1.f + (qlt - p)) * (1.f - (qrb - p)); // g_lt == g_rb in source
        int il = (int)qlt, ir = (int)qrb;
        float gl = (il >= 1 && il <= L_) ? g : 0.f;      // pad tap -> 0
        float gr = (ir >= 1 && ir <= L_) ? g : 0.f;
        il = min(max(il - 1, 0), L_ - 1);                // to unpadded index
        ir = min(max(ir - 1, 0), L_ - 1);
        size_t pi = (((size_t)b * H_ + i) * KS_ + k) * 4;
        params[pi + 0] = gl;
        params[pi + 1] = gr;
        params[pi + 2] = __int_as_float(il);
        params[pi + 3] = __int_as_float(ir);
    }
}

// ---- Kernel 2: fused gather + bf16 WMMA GEMM ----
// One block = one (b, 128-position tile). 8 waves, wave m owns oc 16m..16m+15.
__global__ __launch_bounds__(256) void dconv_main(const float* __restrict__ x,
                                                  const float* __restrict__ params,
                                                  const unsigned short* __restrict__ wp,
                                                  float* __restrict__ out) {
    __shared__ __align__(32) unsigned short Bp[6 * 8 * 32 * 16]; // 48 KB B-panel
    __shared__ float4 pc[128 * KS_];                             // 6 KB params
    int b = blockIdx.x >> 5;
    int tile = blockIdx.x & 31;
    int posBase = tile * 128;
    int tid = threadIdx.x;

    const float4* pg = (const float4*)params;
    size_t pbase = ((size_t)b * H_ + posBase) * KS_;
    for (int q = tid; q < 128 * KS_; q += 256) pc[q] = pg[pbase + q];
    __syncthreads();

    // Gather B-panel into WMMA B-fragment layout.
    // B 32x16 bf16: N = lane&15; element e<8 holds K 8h+e, e>=8 holds
    // K 16+8h+(e-8), h = lane>>4 (same K order as A).
    const float* xb = x + (size_t)b * C_ * L_;
    for (int f = tid; f < 6 * 8 * 32 * 16; f += 256) {
        int e    = f & 15;
        int lane = (f >> 4) & 31;
        int nsub = (f >> 9) & 7;
        int kk   = f >> 12;
        int K = kk * 32 + ((e & 8) ? 16 : 0) + ((lane & 16) ? 8 : 0) + (e & 7);
        int c = K / KS_;
        int kpos = K - c * KS_;
        int lp = nsub * 16 + (lane & 15);
        float4 pr = pc[lp * KS_ + kpos];
        int il = __float_as_int(pr.z);
        int ir = __float_as_int(pr.w);
        float val = pr.x * xb[c * L_ + il] + pr.y * xb[c * L_ + ir];
        Bp[f] = f32_bf16(val);
    }

    // A fragments (weights, L2-hot): 6 K-steps x 16 bf16 per lane.
    int wave = tid >> 5, lane = tid & 31;
    v16bf afr[6];
    #pragma unroll
    for (int kk = 0; kk < 6; ++kk)
        afr[kk] = *(const v16bf*)(wp + (((wave * 6 + kk) * 32) + lane) * 16);
    __syncthreads();

    // GEMM: 8 N-subtiles x 6 K-steps of v_wmma_f32_16x16x32_bf16.
    float* ob = out + (size_t)b * OC_ * H_;
    int ocBase = wave * 16 + ((lane & 16) ? 8 : 0);   // C/D layout: VGPR r -> M=r / 8+r
    int iCol = posBase + (lane & 15);
    #pragma unroll
    for (int nsub = 0; nsub < 8; ++nsub) {
        v8f acc = {};
        #pragma unroll
        for (int kk = 0; kk < 6; ++kk) {
            v16bf bf = *(const v16bf*)(&Bp[((kk * 8 + nsub) * 32 + lane) * 16]);
            acc = __builtin_amdgcn_wmma_f32_16x16x32_bf16(
                false, afr[kk], false, bf, (short)0, acc, false, false);
        }
        int i = iCol + nsub * 16;
        #pragma unroll
        for (int r = 0; r < 8; ++r)
            ob[(size_t)(ocBase + r) * H_ + i] = acc[r];
    }
}

extern "C" void kernel_launch(void* const* d_in, const int* in_sizes, int n_in,
                              void* d_out, int out_size, void* d_ws, size_t ws_size,
                              hipStream_t stream) {
    const float* x      = (const float*)d_in[0];   // (64,64,4096)
    const float* w_off  = (const float*)d_in[1];   // (3,64,3)
    const float* b_off  = (const float*)d_in[2];   // (3,)
    const float* w_conv = (const float*)d_in[3];   // (128,64,3)
    float* out = (float*)d_out;                    // (64,128,4096)

    float* params = (float*)d_ws;                                   // 12.6 MB
    unsigned short* wp = (unsigned short*)((char*)d_ws + PARAM_FLOATS * sizeof(float)); // 48 KB

    pack_w<<<(OC_ * KTOT + 255) / 256, 256, 0, stream>>>(w_conv, wp);
    offs_k<<<B_ * 32, 128, 0, stream>>>(x, w_off, b_off, params);
    dconv_main<<<B_ * 32, 256, 0, stream>>>(x, params, wp, out);
}
